// Injector_51084341019190
// MI455X (gfx1250) — compile-verified, matrix operand order checked by reading, not run on hardware
//
#include <hip/hip_runtime.h>

// ---------------------------------------------------------------------------
// MSDeformAttn block for MI455X (gfx1250, wave32).
// GEMMs via v_wmma_f32_16x16x32_bf16 (bf16 in, f32 acc) with 2x4 (or 2x3)
// register-blocked tiles per wave: A frags reused 4x, B frags 2x ->
// 1.5 b128 loads per WMMA instead of 4. Gather stays fp32; value (67 MB)
// fits in the 192 MB L2 -> gather is L2-bandwidth bound, not HBM bound.
// ---------------------------------------------------------------------------

#define B_   4
#define NQ_  16384
#define NV_  16384
#define D_   256
#define HEADS_ 8
#define HD_  32
#define PTS_ 4
#define HL_  128
#define WL_  128
#define NPROJ_ 96   // 64 offset cols + 32 attn cols

typedef __attribute__((ext_vector_type(16))) __bf16 v16bf;
typedef __attribute__((ext_vector_type(8)))  __bf16 v8bf;
typedef __attribute__((ext_vector_type(8)))  float  v8f;

static __device__ __forceinline__ unsigned short f2bf(float f) {
  unsigned int u = __float_as_uint(f);
  u = u + 0x7FFFu + ((u >> 16) & 1u);     // round-to-nearest-even
  return (unsigned short)(u >> 16);
}

// ---------------------------------------------------------------------------
// LayerNorm over D=256, one wave per row, output bf16 (as ushort bits).
// ---------------------------------------------------------------------------
__global__ void ln_bf16_kernel(const float* __restrict__ x,
                               const float* __restrict__ scale,
                               const float* __restrict__ bias,
                               unsigned short* __restrict__ y, int rows) {
  int row = blockIdx.x * (blockDim.x >> 5) + (threadIdx.x >> 5);
  if (row >= rows) return;
  int lane = threadIdx.x & 31;
  const float* xr = x + (size_t)row * D_;
  float v[8], s = 0.f, ss = 0.f;
#pragma unroll
  for (int j = 0; j < 8; ++j) { v[j] = xr[lane + j * 32]; s += v[j]; ss += v[j] * v[j]; }
#pragma unroll
  for (int o = 16; o > 0; o >>= 1) { s += __shfl_xor(s, o, 32); ss += __shfl_xor(ss, o, 32); }
  float mu  = s * (1.0f / D_);
  float var = ss * (1.0f / D_) - mu * mu;
  float inv = rsqrtf(var + 1e-6f);
  unsigned short* yr = y + (size_t)row * D_;
#pragma unroll
  for (int j = 0; j < 8; ++j) {
    int c = lane + j * 32;
    yr[c] = f2bf((v[j] - mu) * inv * scale[c] + bias[c]);
  }
}

// ---------------------------------------------------------------------------
// Pack a K=256 x N fp32 weight into per-lane WMMA B fragments (bf16).
// dst u32 index t = ((ntile*8 + ks)*32 + lane)*8 + j ; lane holds col
// n = ntile*16 + lane%16, K striping per ISA 16-bit A/B layout.
// ---------------------------------------------------------------------------
static __device__ __forceinline__ int kmap(int lane, int j) {
  return (lane & 16) ? ((j < 4) ? (8 + 2 * j) : (24 + 2 * (j - 4)))
                     : ((j < 4) ? (2 * j)     : (16 + 2 * (j - 4)));
}

__global__ void pack_w_kernel(const float* __restrict__ W, int N,
                              unsigned int* __restrict__ dst) {
  int t = blockIdx.x * blockDim.x + threadIdx.x;
  if (t >= 128 * N) return;
  int j = t & 7, lane = (t >> 3) & 31, ks = (t >> 8) & 7, ntile = t >> 11;
  int klo = ks * 32 + kmap(lane, j);
  int n = ntile * 16 + (lane & 15);
  unsigned int lo = f2bf(W[(size_t)klo * N + n]);
  unsigned int hi = f2bf(W[(size_t)(klo + 1) * N + n]);
  dst[t] = lo | (hi << 16);
}

__global__ void pack_w96_kernel(const float* __restrict__ Woff,
                                const float* __restrict__ Watt,
                                unsigned int* __restrict__ dst) {
  int t = blockIdx.x * blockDim.x + threadIdx.x;
  if (t >= 128 * NPROJ_) return;
  int j = t & 7, lane = (t >> 3) & 31, ks = (t >> 8) & 7, ntile = t >> 11;
  int klo = ks * 32 + kmap(lane, j);
  int n = ntile * 16 + (lane & 15);
  float flo = (n < 64) ? Woff[(size_t)klo * 64 + n]       : Watt[(size_t)klo * 32 + (n - 64)];
  float fhi = (n < 64) ? Woff[(size_t)(klo + 1) * 64 + n] : Watt[(size_t)(klo + 1) * 32 + (n - 64)];
  dst[t] = (unsigned int)f2bf(flo) | ((unsigned int)f2bf(fhi) << 16);
}

// ---------------------------------------------------------------------------
// Register-blocked WMMA GEMM core: one wave computes MT x NT 16x16 tiles,
// K = 256 (8 k-steps, MT*NT v_wmma per step). A frags reused NT times,
// B frags reused MT times.
// ---------------------------------------------------------------------------
union Frag { v16bf v; v8bf h[2]; };

template <int MT, int NT>
__device__ __forceinline__ void wmma_block256(const __bf16* __restrict__ A,
                                              const __bf16* __restrict__ Wp,
                                              int mtile0, int ntile0,
                                              v8f (&c)[MT][NT]) {
  const int lane = threadIdx.x & 31;
  const int ahi = (lane & 16) ? 8 : 0;
  const __bf16* arow[MT];
#pragma unroll
  for (int mt = 0; mt < MT; ++mt)
    arow[mt] = A + (size_t)((mtile0 + mt) * 16 + (lane & 15)) * D_;
  const __bf16* wfrag[NT];
#pragma unroll
  for (int nt = 0; nt < NT; ++nt)
    wfrag[nt] = Wp + ((size_t)((ntile0 + nt) * 8) * 32 + lane) * 16;

#pragma unroll
  for (int ks = 0; ks < 8; ++ks) {
    Frag a[MT], b[NT];
#pragma unroll
    for (int mt = 0; mt < MT; ++mt) {
      const __bf16* ap = arow[mt] + ks * 32 + ahi;
      a[mt].h[0] = *(const v8bf*)(ap);
      a[mt].h[1] = *(const v8bf*)(ap + 16);
    }
#pragma unroll
    for (int nt = 0; nt < NT; ++nt) {
      const __bf16* wp = wfrag[nt] + (size_t)ks * 32 * 16;
      b[nt].h[0] = *(const v8bf*)(wp);
      b[nt].h[1] = *(const v8bf*)(wp + 8);
    }
#pragma unroll
    for (int mt = 0; mt < MT; ++mt)
#pragma unroll
      for (int nt = 0; nt < NT; ++nt)
        c[mt][nt] = __builtin_amdgcn_wmma_f32_16x16x32_bf16(
            false, a[mt].v, false, b[nt].v, (short)0, c[mt][nt], false, false);
  }
}

// value = LN(feat) @ W_value + b_value, scattered to (B, HEADS, NV, HD) fp32
__global__ void value_gemm_kernel(const __bf16* __restrict__ fln,
                                  const __bf16* __restrict__ Wp,
                                  const float* __restrict__ bval,
                                  float* __restrict__ value) {
  int tid = blockIdx.x * (blockDim.x >> 5) + (threadIdx.x >> 5);
  int ngrp = tid & 3, mgrp = tid >> 2;            // 4 x 2048 groups
  int mtile0 = mgrp * 2, ntile0 = ngrp * 4;
  v8f c[2][4] = {};
  wmma_block256<2, 4>(fln, Wp, mtile0, ntile0, c);
  int lane = threadIdx.x & 31;
#pragma unroll
  for (int mt = 0; mt < 2; ++mt) {
    int rbase = (mtile0 + mt) * 16 + ((lane & 16) ? 8 : 0);
#pragma unroll
    for (int nt = 0; nt < 4; ++nt) {
      int col = (ntile0 + nt) * 16 + (lane & 15);
      int head = col >> 5, hd = col & 31;
      float bias = bval[col];
#pragma unroll
      for (int i = 0; i < 8; ++i) {
        int r = rbase + i;
        int b = r >> 14, rl = r & (NV_ - 1);
        value[(((size_t)(b * HEADS_ + head) * NV_ + rl) << 5) + hd] = c[mt][nt][i] + bias;
      }
    }
  }
}

// proj = LN(q) @ [W_off | W_attn] + bias, fp32 (B*NQ, 96)
__global__ void proj_gemm_kernel(const __bf16* __restrict__ qln,
                                 const __bf16* __restrict__ Wp,
                                 const float* __restrict__ boff,
                                 const float* __restrict__ batt,
                                 float* __restrict__ proj) {
  int tid = blockIdx.x * (blockDim.x >> 5) + (threadIdx.x >> 5);
  int ngrp = tid & 1, mgrp = tid >> 1;            // 2 x 2048 groups
  int mtile0 = mgrp * 2, ntile0 = ngrp * 3;
  v8f c[2][3] = {};
  wmma_block256<2, 3>(qln, Wp, mtile0, ntile0, c);
  int lane = threadIdx.x & 31;
#pragma unroll
  for (int mt = 0; mt < 2; ++mt) {
    int rbase = (mtile0 + mt) * 16 + ((lane & 16) ? 8 : 0);
#pragma unroll
    for (int nt = 0; nt < 3; ++nt) {
      int col = (ntile0 + nt) * 16 + (lane & 15);
      float bias = (col < 64) ? boff[col] : batt[col - 64];
#pragma unroll
      for (int i = 0; i < 8; ++i)
        proj[(size_t)(rbase + i) * NPROJ_ + col] = c[mt][nt][i] + bias;
    }
  }
}

// ---------------------------------------------------------------------------
// Bilinear gather: one wave per (b, q, head); lane = channel (HD=32).
// value resides in L2 (67 MB < 192 MB), so the 4pts x 4corners x 128B reads
// are L2-bandwidth bound.
// ---------------------------------------------------------------------------
static __device__ __forceinline__ float corner(const float* __restrict__ vb,
                                               int xi, int yi, float w, int d) {
  bool inb = (xi >= 0) & (xi < WL_) & (yi >= 0) & (yi < HL_);
  int cx = min(max(xi, 0), WL_ - 1);
  int cy = min(max(yi, 0), HL_ - 1);
  float g = vb[(size_t)((cy << 7) + cx) * HD_ + d];
  return inb ? w * g : 0.f;
}

__global__ void gather_kernel(const float* __restrict__ proj,
                              const float* __restrict__ refp,
                              const float* __restrict__ value,
                              unsigned short* __restrict__ gath) {
  int wid = blockIdx.x * (blockDim.x >> 5) + (threadIdx.x >> 5);
  int h = wid & 7, q = (wid >> 3) & (NQ_ - 1), b = wid >> 17;
  int lane = threadIdx.x & 31;
  size_t bq = (size_t)b * NQ_ + q;
  const float* prow = proj + bq * NPROJ_;
  float rx = refp[bq * 2 + 0] * (float)WL_ - 0.5f;
  float ry = refp[bq * 2 + 1] * (float)HL_ - 0.5f;

  // softmax over 4 attention logits
  float a[PTS_], m = -3.4e38f;
#pragma unroll
  for (int p = 0; p < PTS_; ++p) { a[p] = prow[64 + h * 4 + p]; m = fmaxf(m, a[p]); }
  float sum = 0.f;
#pragma unroll
  for (int p = 0; p < PTS_; ++p) { a[p] = expf(a[p] - m); sum += a[p]; }
  float rcp = 1.0f / sum;

  const float* vb = value + (size_t)(b * HEADS_ + h) * NV_ * HD_;
  float acc = 0.f;
#pragma unroll
  for (int p = 0; p < PTS_; ++p) {
    float x = rx + prow[h * 8 + p * 2 + 0];   // ref*W + off  (offset_norm = W)
    float y = ry + prow[h * 8 + p * 2 + 1];
    float x0f = floorf(x), y0f = floorf(y);
    int x0 = (int)x0f, y0 = (int)y0f;
    float fx = x - x0f, fy = y - y0f;
    float s = corner(vb, x0,     y0,     (1.f - fx) * (1.f - fy), lane)
            + corner(vb, x0 + 1, y0,     fx         * (1.f - fy), lane)
            + corner(vb, x0,     y0 + 1, (1.f - fx) * fy,         lane)
            + corner(vb, x0 + 1, y0 + 1, fx         * fy,         lane);
    acc += (a[p] * rcp) * s;
  }
  gath[bq * D_ + h * HD_ + lane] = f2bf(acc);
}

// out = query + gamma * (gath @ W_out + b_out)
__global__ void out_gemm_kernel(const __bf16* __restrict__ gath,
                                const __bf16* __restrict__ Wp,
                                const float* __restrict__ bout,
                                const float* __restrict__ query,
                                const float* __restrict__ gamma,
                                float* __restrict__ out) {
  int tid = blockIdx.x * (blockDim.x >> 5) + (threadIdx.x >> 5);
  int ngrp = tid & 3, mgrp = tid >> 2;            // 4 x 2048 groups
  int mtile0 = mgrp * 2, ntile0 = ngrp * 4;
  v8f c[2][4] = {};
  wmma_block256<2, 4>(gath, Wp, mtile0, ntile0, c);
  int lane = threadIdx.x & 31;
#pragma unroll
  for (int mt = 0; mt < 2; ++mt) {
    int rbase = (mtile0 + mt) * 16 + ((lane & 16) ? 8 : 0);
#pragma unroll
    for (int nt = 0; nt < 4; ++nt) {
      int col = (ntile0 + nt) * 16 + (lane & 15);
      float bias = bout[col], g = gamma[col];
#pragma unroll
      for (int i = 0; i < 8; ++i) {
        size_t idx = (size_t)(rbase + i) * D_ + col;
        out[idx] = query[idx] + g * (c[mt][nt][i] + bias);
      }
    }
  }
}

// ---------------------------------------------------------------------------
extern "C" void kernel_launch(void* const* d_in, const int* in_sizes, int n_in,
                              void* d_out, int out_size, void* d_ws, size_t ws_size,
                              hipStream_t stream) {
  const float* query = (const float*)d_in[0];
  const float* feat  = (const float*)d_in[1];
  const float* refp  = (const float*)d_in[2];
  const float* qn_s  = (const float*)d_in[5];
  const float* qn_b  = (const float*)d_in[6];
  const float* fn_s  = (const float*)d_in[7];
  const float* fn_b  = (const float*)d_in[8];
  const float* Wv    = (const float*)d_in[9];
  const float* bV    = (const float*)d_in[10];
  const float* Woff  = (const float*)d_in[11];
  const float* boff  = (const float*)d_in[12];
  const float* Watt  = (const float*)d_in[13];
  const float* batt  = (const float*)d_in[14];
  const float* Wout  = (const float*)d_in[15];
  const float* bout  = (const float*)d_in[16];
  const float* gamma = (const float*)d_in[17];

  char* ws = (char*)d_ws;
  const size_t ROWS = (size_t)B_ * NQ_;                    // 65536
  unsigned short* q_ln = (unsigned short*)(ws);            size_t o = ROWS * D_ * 2;
  unsigned short* f_ln = (unsigned short*)(ws + o);        o += ROWS * D_ * 2;
  unsigned int*   wv_p = (unsigned int*)(ws + o);          o += (size_t)D_ * D_ * 2;
  unsigned int*   wq_p = (unsigned int*)(ws + o);          o += (size_t)D_ * NPROJ_ * 2;
  unsigned int*   wo_p = (unsigned int*)(ws + o);          o += (size_t)D_ * D_ * 2;
  float*          valv = (float*)(ws + o);                 o += ROWS * D_ * 4;
  float*          proj = (float*)(ws + o);                 o += ROWS * NPROJ_ * 4;
  unsigned short* gath = (unsigned short*)(ws + o);

  // 1. LayerNorms -> bf16 (8 rows / 256-thread block)
  ln_bf16_kernel<<<ROWS / 8, 256, 0, stream>>>(query, qn_s, qn_b, q_ln, (int)ROWS);
  ln_bf16_kernel<<<ROWS / 8, 256, 0, stream>>>(feat,  fn_s, fn_b, f_ln, (int)ROWS);

  // 2. Pack weights into WMMA B-fragment layout
  pack_w_kernel  <<<(128 * D_ + 255) / 256, 256, 0, stream>>>(Wv,   D_, wv_p);
  pack_w96_kernel<<<(128 * NPROJ_ + 255) / 256, 256, 0, stream>>>(Woff, Watt, wq_p);
  pack_w_kernel  <<<(128 * D_ + 255) / 256, 256, 0, stream>>>(Wout, D_, wo_p);

  // 3. value GEMM: 2048 x 4 groups (each wave: 2x4 tiles), 8 waves/block
  value_gemm_kernel<<<(2048 * 4) / 8, 256, 0, stream>>>(
      (const __bf16*)f_ln, (const __bf16*)wv_p, bV, valv);

  // 4. offset/attn GEMM: 2048 x 2 groups (each wave: 2x3 tiles)
  proj_gemm_kernel<<<(2048 * 2) / 8, 256, 0, stream>>>(
      (const __bf16*)q_ln, (const __bf16*)wq_p, boff, batt, proj);

  // 5. bilinear gather: one wave per (b, q, head)
  gather_kernel<<<(B_ * NQ_ * HEADS_) / 8, 256, 0, stream>>>(proj, refp, valv, gath);

  // 6. output GEMM + residual
  out_gemm_kernel<<<(2048 * 4) / 8, 256, 0, stream>>>(
      (const __bf16*)gath, (const __bf16*)wo_p, bout, query, gamma, (float*)d_out);
}
